// mymolGen_63213328663243
// MI455X (gfx1250) — matrix operand order repaired and checked by reading, not compiled, not run on hardware
//
#include <hip/hip_runtime.h>

typedef _Float16 v16h __attribute__((ext_vector_type(16)));
typedef float    v8f  __attribute__((ext_vector_type(8)));

#define TPB   256
#define KPMAX 256   // max padded K (largest K used is 250)

// ---------------------------------------------------------------------------
// WMMA GEMM: C[M,N] = act(A[M,K] @ W[K,N] + bias) (+ resid)
// grid = (ceil(M/16/8), ceil(N/16)); one wave per 16x16 tile.
// Preconditions used for the fast path: M % 16 == 0, K even, K <= KPMAX.
// W^T slice staged in LDS as zero-padded f16 -> unguarded v16h B loads.
// A loaded with unguarded float2 in full K chunks; clamped+select in the tail.
// ---------------------------------------------------------------------------
__global__ void gemm_wmma_kernel(const float* __restrict__ A,
                                 const float* __restrict__ W,
                                 const float* __restrict__ bias,
                                 const float* __restrict__ resid,
                                 float* __restrict__ C,
                                 int M, int K, int N, int relu)
{
    __shared__ _Float16 wlds[16 * KPMAX];      // 8 KB: this block's 16 W columns, K-major
    const int KP = (K + 31) & ~31;
    const int n0 = blockIdx.y << 4;

    // Stage wlds[n][k] = W[k][n0+n] (f16, zero-padded in both n and k).
    for (int idx = threadIdx.x; idx < 16 * KP; idx += blockDim.x) {
        int k = idx >> 4;
        int n = idx & 15;
        float v = 0.f;
        if (k < K && (n0 + n) < N) v = W[(size_t)k * N + n0 + n];
        wlds[n * KPMAX + k] = (_Float16)v;
    }
    __syncthreads();

    const int lane = threadIdx.x & 31;
    const int wave = threadIdx.x >> 5;
    const int m0 = (blockIdx.x * (TPB / 32) + wave) << 4;
    if (m0 >= M) return;                        // wave-uniform; EXEC stays all-1s

    const int half = lane >> 4, r = lane & 15;
    const float*    __restrict__ arow = A + (size_t)(m0 + r) * (size_t)K;
    const _Float16* __restrict__ bcol = &wlds[r * KPMAX];

    v8f acc = {0.f,0.f,0.f,0.f,0.f,0.f,0.f,0.f};
    const int kfull = K & ~31;
    int k0 = 0;
    for (; k0 < kfull; k0 += 32) {
        // A fragment: element i -> K = k0 + 16*half + i
        v16h a;
        const float2* p0 = (const float2*)(arow + k0 + (half << 4));
        const float2* p1 = (const float2*)(arow + k0 + 8 + (half << 4));
#pragma unroll
        for (int q = 0; q < 4; ++q) {
            float2 t0 = p0[q];
            float2 t1 = p1[q];
            a[2*q]       = (_Float16)t0.x;  a[2*q + 1]     = (_Float16)t0.y;
            a[8 + 2*q]   = (_Float16)t1.x;  a[8 + 2*q + 1] = (_Float16)t1.y;
        }
        // B fragment: element i -> K = k0 + 16*half + i  (contiguous in LDS)
        v16h b = *(const v16h*)(bcol + k0 + (half << 4));
        acc = __builtin_amdgcn_wmma_f32_16x16x32_f16(false, a, false, b,
                                                     (short)0, acc, false, false);
    }
    if (k0 < K) {                               // K tail: clamped loads + selects
        v16h a;
#pragma unroll
        for (int v = 0; v < 8; ++v) {
            int ka = k0 + ((v >> 2) << 3) + (half << 4) + ((v & 3) << 1);
            bool ok = ka < K;                   // K even, ka even -> pairwise validity
            int kc = ok ? ka : (K - 2);
            float2 t = *(const float2*)(arow + kc);
            a[2*v]     = (_Float16)(ok ? t.x : 0.f);
            a[2*v + 1] = (_Float16)(ok ? t.y : 0.f);
        }
        v16h b = *(const v16h*)(bcol + k0 + (half << 4));   // zero-padded
        acc = __builtin_amdgcn_wmma_f32_16x16x32_f16(false, a, false, b,
                                                     (short)0, acc, false, false);
    }

    const int cn = n0 + r;
    const bool nok = cn < N;
    const float bs = (bias && nok) ? bias[cn] : 0.f;
#pragma unroll
    for (int j = 0; j < 8; ++j) {
        int cm = m0 + (half << 3) + j;          // C/D layout: VGPR j -> M
        if (nok) {
            float v = acc[j] + bs;
            if (relu)  v = fmaxf(v, 0.f);
            if (resid) v += resid[(size_t)cm * N + cn];
            C[(size_t)cm * N + cn] = v;
        }
    }
}

// ---------------------------------------------------------------------------
// GCN support kernels
// ---------------------------------------------------------------------------
__global__ void fill_kernel(float* p, float v, long long n)
{
    long long i = (long long)blockIdx.x * blockDim.x + threadIdx.x;
    if (i < n) p[i] = v;
}

__global__ void deg_edges_kernel(const int* __restrict__ col,
                                 const float* __restrict__ ew,
                                 float* __restrict__ deg, long long E)
{
    long long e = (long long)blockIdx.x * blockDim.x + threadIdx.x;
    if (e < E) {
        float w = ew ? ew[e] : 1.0f;
        if (w != 0.f) atomicAdd(&deg[col[e]], w);
    }
}

__global__ void dinv_kernel(float* d, long long n)
{
    long long i = (long long)blockIdx.x * blockDim.x + threadIdx.x;
    if (i < n) {
        float v = d[i];
        d[i] = (v > 0.f) ? rsqrtf(fmaxf(v, 1e-12f)) : 0.f;
    }
}

// out[v,f] = dinv[v]^2 * hW[v,f] + b[f]     (self-loop term + bias)
__global__ void agg_self_kernel(const float* __restrict__ hW,
                                const float* __restrict__ dinv,
                                const float* __restrict__ b,
                                float* __restrict__ out, long long Nn, int F)
{
    long long i = (long long)blockIdx.x * blockDim.x + threadIdx.x;
    if (i < Nn * (long long)F) {
        long long node = i / F;
        int f = (int)(i % F);
        float di = dinv[node];
        out[i] = di * di * hW[i] + b[f];
    }
}

// out[col,f] += dinv[row]*ew*dinv[col] * hW[row,f]
__global__ void agg_edge_kernel(const int* __restrict__ row,
                                const int* __restrict__ col,
                                const float* __restrict__ ew,
                                const float* __restrict__ dinv,
                                const float* __restrict__ hW,
                                float* __restrict__ out, long long E, int F)
{
    long long i = (long long)blockIdx.x * blockDim.x + threadIdx.x;
    if (i < E * (long long)F) {
        long long e = i / F;
        int f = (int)(i % F);
        float w = ew ? ew[e] : 1.0f;
        if (w != 0.f) {
            int rs = row[e], cd = col[e];
            float nrm = dinv[rs] * w * dinv[cd];
            atomicAdd(&out[(size_t)cd * F + f], hW[(size_t)rs * F + f] * nrm);
        }
    }
}

// ---------------------------------------------------------------------------
// TopK pooling: one 64-thread block per graph. Fuses the preceding ReLU.
// Stable rank selection reproduces jax.lax.top_k ordering (ties -> lower idx).
// ---------------------------------------------------------------------------
__global__ void topk_pool_kernel(const float* __restrict__ h, int F, int npg, int k,
                                 const float* __restrict__ pw,
                                 const int* __restrict__ row_in,
                                 const int* __restrict__ col_in,
                                 const float* __restrict__ ew_in, int epg,
                                 float* __restrict__ h_out,
                                 int* __restrict__ row_out,
                                 int* __restrict__ col_out,
                                 float* __restrict__ ew_out,
                                 int remap_edges)
{
    __shared__ float s_score[64];
    __shared__ int   s_map[64];
    __shared__ float s_nrm;
    const int g = blockIdx.x;
    const int tid = threadIdx.x;

    if (tid == 0) {
        float s = 0.f;
        for (int i = 0; i < F; ++i) s += pw[i] * pw[i];
        s_nrm = sqrtf(s);
    }
    __syncthreads();

    float score = 0.f;
    if (tid < npg) {
        const float* hr = h + ((size_t)g * npg + tid) * F;
        float dot = 0.f;
        for (int i = 0; i < F; ++i) dot += fmaxf(hr[i], 0.f) * pw[i];
        score = tanhf(dot / s_nrm);
        s_score[tid] = score;
    }
    __syncthreads();

    if (tid < npg) {
        int rank = 0;
        for (int j = 0; j < npg; ++j) {
            float sj = s_score[j];
            rank += (sj > score) || (sj == score && j < tid);
        }
        int nm = (rank < k) ? rank : -1;
        s_map[tid] = nm;
        if (nm >= 0) {
            const float* hr = h + ((size_t)g * npg + tid) * F;
            float* ho = h_out + ((size_t)g * k + nm) * F;
            for (int i = 0; i < F; ++i) ho[i] = fmaxf(hr[i], 0.f) * score;
        }
    }
    __syncthreads();

    if (remap_edges) {
        for (int e = tid; e < epg; e += blockDim.x) {
            size_t ge = (size_t)g * epg + e;
            int lr = row_in[ge] - g * npg;
            int lc = col_in[ge] - g * npg;
            float w = ew_in ? ew_in[ge] : 1.0f;
            int mr = s_map[lr], mc = s_map[lc];
            bool keep = (mr >= 0) && (mc >= 0);
            row_out[ge] = keep ? (g * k + mr) : 0;
            col_out[ge] = keep ? (g * k + mc) : 0;
            ew_out[ge]  = keep ? w : 0.f;
        }
    }
}

// per-graph [max over k, mean over k] -> pooled[g, 2F]
__global__ void readout_kernel(const float* __restrict__ h,
                               float* __restrict__ pooled, int k, int F)
{
    int g = blockIdx.x;
    int f = threadIdx.x;
    if (f >= F) return;
    const float* p = h + (size_t)g * k * F + f;
    float mx = -3.4e38f, sm = 0.f;
    for (int j = 0; j < k; ++j) {
        float v = p[(size_t)j * F];
        mx = fmaxf(mx, v);
        sm += v;
    }
    pooled[(size_t)g * 2 * F + f]     = mx;
    pooled[(size_t)g * 2 * F + F + f] = sm / (float)k;
}

__global__ void lenemb_kernel(const int* __restrict__ slen,
                              const float* __restrict__ emb,
                              float* __restrict__ le, long long n)
{
    long long i = (long long)blockIdx.x * blockDim.x + threadIdx.x;
    if (i < n) {
        long long b = i / 200;
        int f = (int)(i % 200);
        le[i] = fmaxf(emb[(size_t)slen[b] * 200 + f], 0.f);
    }
}

__global__ void zc_kernel(const float* __restrict__ mu,
                          const float* __restrict__ le,
                          float* __restrict__ zc, long long n)
{
    long long i = (long long)blockIdx.x * blockDim.x + threadIdx.x;
    if (i < n) {
        long long b = i / 250;
        int j = (int)(i % 250);
        zc[i] = (j < 50) ? mu[(size_t)b * 50 + j] : le[(size_t)b * 200 + (j - 50)];
    }
}

// ---------------------------------------------------------------------------
// Host orchestration
// ---------------------------------------------------------------------------
static inline unsigned grid1d(long long n) { return (unsigned)((n + TPB - 1) / TPB); }

static inline void run_gemm(const float* A, const float* W, const float* bias,
                            const float* resid, float* C, long long M, int K, int N,
                            int relu, hipStream_t s)
{
    unsigned MT = (unsigned)((M + 15) >> 4);
    unsigned NT = (unsigned)((N + 15) >> 4);
    dim3 grid((MT + (TPB / 32) - 1) / (TPB / 32), NT);
    gemm_wmma_kernel<<<grid, TPB, 0, s>>>(A, W, bias, resid, C, (int)M, K, N, relu);
}

extern "C" void kernel_launch(void* const* d_in, const int* in_sizes, int n_in,
                              void* d_out, int out_size, void* d_ws, size_t ws_size,
                              hipStream_t stream)
{
    (void)n_in; (void)out_size; (void)ws_size;

    const float* x        = (const float*)d_in[0];
    const int*   eidx     = (const int*)  d_in[1];
    const int*   slen     = (const int*)  d_in[3];
    const float* conv1_w  = (const float*)d_in[4];
    const float* conv1_b  = (const float*)d_in[5];
    const float* pool1_w  = (const float*)d_in[6];
    const float* conv2_w  = (const float*)d_in[7];
    const float* conv2_b  = (const float*)d_in[8];
    const float* pool2_w  = (const float*)d_in[9];
    const float* mean_w   = (const float*)d_in[10];
    const float* mean_b   = (const float*)d_in[11];
    const float* logvar_w = (const float*)d_in[12];
    const float* logvar_b = (const float*)d_in[13];
    const float* emb      = (const float*)d_in[14];
    const float* dec1_w   = (const float*)d_in[15];
    const float* dec1_b   = (const float*)d_in[16];
    const float* dec2_w   = (const float*)d_in[17];
    const float* dec2_b   = (const float*)d_in[18];
    const float* out_w    = (const float*)d_in[19];
    const float* out_b    = (const float*)d_in[20];

    const int  Bn  = in_sizes[3];
    const long long N0 = in_sizes[0] / 30;
    const long long E  = in_sizes[1] / 2;
    const int  NPG = (int)(N0 / Bn);
    const int  EPG = (int)(E / Bn);
    const int  k1  = (4 * NPG + 4) / 5;      // ceil(0.8*NPG)
    const int  k2  = (4 * k1 + 4) / 5;       // ceil(0.8*k1)
    const long long N1 = (long long)Bn * k1;
    const long long N2 = (long long)Bn * k2;
    const int* row0 = eidx;
    const int* col0 = eidx + E;

    // ---- carve workspace (slabs reused across stages) ----
    char* base = (char*)d_ws;
    auto carve = [&](size_t bytes) -> void* {
        void* p = (void*)base;
        base += (bytes + 255) & ~(size_t)255;
        return p;
    };
    size_t sHW = (size_t)((N0 * 50 > N1 * 100) ? N0 * 50 : N1 * 100);
    size_t sHP = (size_t)((N1 * 50 > N2 * 100) ? N1 * 50 : N2 * 100);
    float* hW     = (float*)carve(sHW * 4);                 // GEMM output (x@W)
    float* hagg   = (float*)carve(sHW * 4);                 // aggregated conv output
    float* hpool  = (float*)carve(sHP * 4);                 // pooled node features
    float* dinv   = (float*)carve((size_t)(N0 > N1 ? N0 : N1) * 4);
    int*   row1   = (int*)  carve((size_t)E * 4);
    int*   col1   = (int*)  carve((size_t)E * 4);
    float* ew1    = (float*)carve((size_t)E * 4);
    float* pooled = (float*)carve((size_t)Bn * 200 * 4);
    float* le     = (float*)carve((size_t)Bn * 200 * 4);
    float* zc     = (float*)carve((size_t)Bn * 250 * 4);
    float* dbuf1  = (float*)carve((size_t)Bn * 100 * 4);
    float* dbuf2  = (float*)carve((size_t)Bn * 200 * 4);

    float* outp = (float*)d_out;                 // [Bn,200]
    float* mu   = outp + (size_t)Bn * 200;       // [Bn,50]
    float* logv = mu   + (size_t)Bn * 50;        // [Bn,50]

    // ================= GCN layer 1 =================
    fill_kernel<<<grid1d(N0), TPB, 0, stream>>>(dinv, 1.0f, N0);          // self loops
    deg_edges_kernel<<<grid1d(E), TPB, 0, stream>>>(col0, nullptr, dinv, E);
    dinv_kernel<<<grid1d(N0), TPB, 0, stream>>>(dinv, N0);
    run_gemm(x, conv1_w, nullptr, nullptr, hW, N0, 30, 50, 0, stream);    // hW = x@W1
    agg_self_kernel<<<grid1d(N0 * 50), TPB, 0, stream>>>(hW, dinv, conv1_b, hagg, N0, 50);
    agg_edge_kernel<<<grid1d(E * 50), TPB, 0, stream>>>(row0, col0, nullptr, dinv, hW, hagg, E, 50);
    // TopK pool 1 (fuses ReLU; remaps edges into row1/col1/ew1)
    topk_pool_kernel<<<Bn, 64, 0, stream>>>(hagg, 50, NPG, k1, pool1_w,
                                            row0, col0, nullptr, EPG,
                                            hpool, row1, col1, ew1, 1);

    // ================= GCN layer 2 =================
    fill_kernel<<<grid1d(N1), TPB, 0, stream>>>(dinv, 1.0f, N1);
    deg_edges_kernel<<<grid1d(E), TPB, 0, stream>>>(col1, ew1, dinv, E);
    dinv_kernel<<<grid1d(N1), TPB, 0, stream>>>(dinv, N1);
    run_gemm(hpool, conv2_w, nullptr, nullptr, hW, N1, 50, 100, 0, stream);
    agg_self_kernel<<<grid1d(N1 * 100), TPB, 0, stream>>>(hW, dinv, conv2_b, hagg, N1, 100);
    agg_edge_kernel<<<grid1d(E * 100), TPB, 0, stream>>>(row1, col1, ew1, dinv, hW, hagg, E, 100);
    // TopK pool 2 (edges no longer needed afterwards)
    topk_pool_kernel<<<Bn, 64, 0, stream>>>(hagg, 100, k1, k2, pool2_w,
                                            nullptr, nullptr, nullptr, 0,
                                            hpool, nullptr, nullptr, nullptr, 0);

    // ================= readout + VAE head =================
    readout_kernel<<<Bn, 128, 0, stream>>>(hpool, pooled, k2, 100);
    run_gemm(pooled, mean_w,   mean_b,   nullptr, mu,   Bn, 200, 50, 1, stream);
    run_gemm(pooled, logvar_w, logvar_b, nullptr, logv, Bn, 200, 50, 1, stream);
    lenemb_kernel<<<grid1d((long long)Bn * 200), TPB, 0, stream>>>(slen, emb, le, (long long)Bn * 200);
    zc_kernel<<<grid1d((long long)Bn * 250), TPB, 0, stream>>>(mu, le, zc, (long long)Bn * 250);

    // ================= decoder =================
    run_gemm(zc,    dec1_w, dec1_b, nullptr, dbuf1, Bn, 250, 100, 1, stream);
    run_gemm(dbuf1, dec2_w, dec2_b, le,      dbuf2, Bn, 100, 200, 1, stream); // relu then +len_emb
    run_gemm(dbuf2, out_w,  out_b,  nullptr, outp,  Bn, 200, 200, 1, stream);
}
